// NodeShuffle_44873818309134
// MI455X (gfx1250) — matrix operand order repaired and checked by reading, not compiled, not run on hardware
//
#include <hip/hip_runtime.h>
#include <hip/hip_bf16.h>

// ---------------------------------------------------------------------------
// NodeShuffle (DGCNN edge-conv + shuffle) for MI455X / gfx1250, wave32 + WMMA.
//   B=4, C=128, N=4096, K=16, SCALE=2
// ---------------------------------------------------------------------------

typedef __attribute__((ext_vector_type(16))) _Float16 v16h;
typedef __attribute__((ext_vector_type(8)))  _Float16 v8h;
typedef __attribute__((ext_vector_type(8)))  float    v8f;
typedef __attribute__((ext_vector_type(2)))  float    v2f;

#define BD   4
#define CD   128
#define ND   4096
#define KNN  16

union V16 { v16h v; v8h h[2]; };

__device__ __forceinline__ v8f wmma_f16(v16h a, v16h b, v8f c) {
  // D = A(16x32 f16) * B(32x16 f16) + C(16x16 f32)
  return __builtin_amdgcn_wmma_f32_16x16x32_f16(false, a, false, b, (short)0, c,
                                                false, false);
}

#if defined(__has_builtin)
#if __has_builtin(__builtin_amdgcn_wmma_f32_16x16x4_f32)
#define HAVE_F32X4 1
__device__ __forceinline__ v8f wmma_f32x4(v2f a, v2f b, v8f c) {
  // D = A(16x4 f32) * B(4x16 f32) + C(16x16 f32)
  return __builtin_amdgcn_wmma_f32_16x16x4_f32(false, a, false, b, (short)0, c,
                                               false, false);
}
#endif
#endif

// ---------------------------------------------------------------------------
// Weight repack: f32 [O,I] row-major -> per-lane WMMA B fragments (f16).
// B-fragment layout (16-bit, 32x16): lane l -> column n = (l&15);
//   element e -> k = ks*32 + 8*(l>=16) + (e&7) + 16*(e>>3)  (table order).
// ---------------------------------------------------------------------------
__device__ __forceinline__ void pack_one(const float* w, _Float16* dst, int id,
                                         int NT, int I) {
  int e    = id & 15;
  int lane = (id >> 4) & 31;
  int rem  = id >> 9;
  int nt   = rem % NT;
  int ks   = rem / NT;
  int co   = nt * 16 + (lane & 15);
  int ci   = ks * 32 + 8 * (lane >> 4) + (e & 7) + 16 * (e >> 3);
  dst[id]  = (_Float16)w[co * I + ci];
}

__global__ void pack_weights_kernel(const float* w1, const float* w2,
                                    const float* w3, const float* wm,
                                    _Float16* w1p, _Float16* w2p,
                                    _Float16* w3p, _Float16* wmp) {
  int id = blockIdx.x * blockDim.x + threadIdx.x;
  if      (id < 32768)  pack_one(w1, w1p, id,          8, 256);  // 128x256
  else if (id < 49152)  pack_one(w2, w2p, id - 32768,  8, 128);  // 128x128
  else if (id < 81920)  pack_one(w3, w3p, id - 49152, 16, 128);  // 256x128
  else if (id < 147456) pack_one(wm, wmp, id - 81920, 16, 256);  // 256x256
}

// Fold eval-mode BN + conv bias into per-channel scale/shift.
__global__ void pack_bn_kernel(const float* b1, const float* g1, const float* be1,
                               const float* m1, const float* v1,
                               const float* b2, const float* g2, const float* be2,
                               const float* m2, const float* v2,
                               float* s1, float* t1, float* s2, float* t2) {
  int c = threadIdx.x;
  if (c < 128) {
    float sa = g1[c] * rsqrtf(v1[c] + 1e-5f);
    s1[c] = sa;
    t1[c] = (b1[c] - m1[c]) * sa + be1[c];
    float sb = g2[c] * rsqrtf(v2[c] + 1e-5f);
    s2[c] = sb;
    t2[c] = (b2[c] - m2[c]) * sb + be2[c];
  }
}

// x[B,C,N] -> p[B,N,C] (f32) and ph (f16)
__global__ void transpose_kernel(const float* __restrict__ x,
                                 float* __restrict__ p,
                                 _Float16* __restrict__ ph) {
  int n = blockIdx.x * blockDim.x + threadIdx.x;
  int c = blockIdx.y;
  int b = blockIdx.z;
  float v = x[((size_t)b * CD + c) * ND + n];
  size_t o = ((size_t)b * ND + n) * CD + c;
  p[o]  = v;
  ph[o] = (_Float16)v;
}

// sq[b,n] = sum_c p^2 : one wave per point.
__global__ void sqnorm_kernel(const float* __restrict__ p, float* __restrict__ sq) {
  int wave = blockIdx.x * (blockDim.x >> 5) + (threadIdx.x >> 5);
  int lane = threadIdx.x & 31;
  const float* row = p + (size_t)wave * CD;
  float s = 0.f;
  #pragma unroll
  for (int c = 0; c < CD; c += 32) { float v = row[c + lane]; s += v * v; }
  #pragma unroll
  for (int d = 16; d; d >>= 1) s += __shfl_xor(s, d, 32);
  if (lane == 0) sq[wave] = s;
}

#if HAVE_F32X4
// Load 16 consecutive f32-WMMA B half-tile fragments (k-steps s0..s0+15).
__device__ __forceinline__ void load_bhalf(v2f (&buf)[16], const float* brow,
                                           int s0, int hi) {
  #pragma unroll
  for (int s = 0; s < 16; ++s)
    buf[s] = *(const v2f*)(brow + (s0 + s) * 4 + 2 * hi);
}

__device__ __forceinline__ v8f chain_half(const v2f* a, const v2f (&bb)[16],
                                          v8f acc) {
  #pragma unroll
  for (int s = 0; s < 16; ++s) acc = wmma_f32x4(a[s], bb[s], acc);
  return acc;
}
#endif

// ---------------------------------------------------------------------------
// kNN: one wave per 16-row tile. dist-proxy = sq_m - 2 * <p_n, p_m>
// (row-constant sq_n dropped: does not change per-row ranking).
// f32 WMMA GEMM with double-buffered (software-pipelined) B half-tiles so
// global loads overlap the WMMA chain instead of serializing on loadcnt.
// Top-16: all 32 lanes scan (2 lanes per row, 8 columns each, private lists
// in LDS); the two per-row lists are merged once at the end. Unsorted is fine:
// the downstream max-over-K is permutation invariant.
// ---------------------------------------------------------------------------
__global__ __launch_bounds__(32) void knn_kernel(const float* __restrict__ p,
                                                 const _Float16* __restrict__ ph,
                                                 const float* __restrict__ sq,
                                                 int* __restrict__ kidx) {
  __shared__ float dls[256];
  __shared__ float bestv[32][16];
  __shared__ int   besti[32][16];

  int lane = threadIdx.x;
  int q  = lane & 15;
  int hi = lane >> 4;
  int n0 = blockIdx.x * 16;
  int b  = blockIdx.y;

  float wval = 3.0e38f;  // current worst (max) in this lane's top-16 list
  int   wslot = 0;
  for (int j = 0; j < 16; ++j) { bestv[lane][j] = 3.0e38f; besti[lane][j] = 0; }
  __syncthreads();

#if HAVE_F32X4
  // f32 A fragments (16x4 per step): lane row M=q, VGPR k = 4s + 2*hi + {0,1}
  const float* arow = p + ((size_t)b * ND + n0 + q) * CD;
  v2f afr[32];
  #pragma unroll
  for (int s = 0; s < 32; ++s) afr[s] = *(const v2f*)(arow + s * 4 + 2 * hi);

  const float* pbase = p + (size_t)b * ND * CD;
  v2f buf0[16], buf1[16];
  load_bhalf(buf0, pbase + (size_t)(0 + q) * CD, 0, hi);   // tile 0, half 0

  for (int mt = 0; mt < ND / 16; ++mt) {
    int m0 = mt * 16;
    int m0n = (mt + 1 < ND / 16) ? (m0 + 16) : m0;          // clamp
    const float* brow  = pbase + (size_t)(m0 + q) * CD;
    const float* brown = pbase + (size_t)(m0n + q) * CD;
    __builtin_prefetch((const void*)(brown + CD), 0, 3);

    load_bhalf(buf1, brow, 16, hi);            // this tile, half 1 (in flight)
    v8f acc = {};
    acc = chain_half(&afr[0], buf0, acc);      // consume half 0
    load_bhalf(buf0, brown, 0, hi);            // next tile, half 0 (in flight)
    acc = chain_half(&afr[16], buf1, acc);     // consume half 1

    // D layout: lane<16: (M=i, N=lane); lane>=16: (M=8+i, N=lane-16)
    float sqm = sq[b * ND + m0 + q];
    #pragma unroll
    for (int i = 0; i < 8; ++i)
      dls[(i + 8 * hi) * 16 + q] = sqm - 2.0f * acc[i];
    __syncthreads();

    // 32-lane scan: lane l handles row (l&15), columns 8*hi .. 8*hi+7
    int cb = 8 * hi;
    for (int cc = 0; cc < 8; ++cc) {
      float d = dls[q * 16 + cb + cc];
      if (d < wval) {
        bestv[lane][wslot] = d;
        besti[lane][wslot] = m0 + cb + cc;
        float wv = -3.0e38f; int wsl = 0;
        for (int j = 0; j < 16; ++j) {
          float bj = bestv[lane][j];
          if (bj > wv) { wv = bj; wsl = j; }
        }
        wval = wv; wslot = wsl;
      }
    }
    __syncthreads();
  }

  // Merge the two 16-lists per row: select 16 smallest of 32.
  if (lane < 16) {
    int r = lane;
    unsigned used = 0u;
    for (int j = 0; j < 16; ++j) {
      float mv = 3.0e38f; int ms = 0;
      for (int t = 0; t < 32; ++t) {
        float v = (t < 16) ? bestv[r][t] : bestv[r + 16][t - 16];
        if (!(used & (1u << t)) && v < mv) { mv = v; ms = t; }
      }
      used |= (1u << ms);
      int mi = (ms < 16) ? besti[r][ms] : besti[r + 16][ms - 16];
      kidx[((size_t)b * ND + n0 + r) * KNN + j] = mi;
    }
  }
#else
  // f16 fallback (not expected to be used on gfx1250)
  const _Float16* arow = ph + ((size_t)b * ND + n0 + q) * CD;
  V16 afr[4];
  #pragma unroll
  for (int ks = 0; ks < 4; ++ks) {
    int c0 = ks * 32 + 16 * hi;
    afr[ks].h[0] = *(const v8h*)(arow + c0);
    afr[ks].h[1] = *(const v8h*)(arow + c0 + 8);
  }
  for (int mt = 0; mt < ND / 16; ++mt) {
    int m0 = mt * 16;
    const _Float16* brow = ph + ((size_t)b * ND + m0 + q) * CD;
    v8f acc = {};
    #pragma unroll
    for (int ks = 0; ks < 4; ++ks) {
      v16h bf = *(const v16h*)(brow + ks * 32 + 16 * hi);
      acc = wmma_f16(afr[ks].v, bf, acc);
    }
    float sqm = sq[b * ND + m0 + q];
    #pragma unroll
    for (int i = 0; i < 8; ++i)
      dls[(i + 8 * hi) * 16 + q] = sqm - 2.0f * acc[i];
    __syncthreads();
    int cb = 8 * hi;
    for (int cc = 0; cc < 8; ++cc) {
      float d = dls[q * 16 + cb + cc];
      if (d < wval) {
        bestv[lane][wslot] = d;
        besti[lane][wslot] = m0 + cb + cc;
        float wv = -3.0e38f; int wsl = 0;
        for (int j = 0; j < 16; ++j) {
          float bj = bestv[lane][j];
          if (bj > wv) { wv = bj; wsl = j; }
        }
        wval = wv; wslot = wsl;
      }
    }
    __syncthreads();
  }
  if (lane < 16) {
    int r = lane;
    unsigned used = 0u;
    for (int j = 0; j < 16; ++j) {
      float mv = 3.0e38f; int ms = 0;
      for (int t = 0; t < 32; ++t) {
        float v = (t < 16) ? bestv[r][t] : bestv[r + 16][t - 16];
        if (!(used & (1u << t)) && v < mv) { mv = v; ms = t; }
      }
      used |= (1u << ms);
      int mi = (ms < 16) ? besti[r][ms] : besti[r + 16][ms - 16];
      kidx[((size_t)b * ND + n0 + r) * KNN + j] = mi;
    }
  }
#endif
}

// ---------------------------------------------------------------------------
// Fused EdgeConv: one wave per point. A = [16 neighbors x 256] edge features
// built directly into WMMA A fragments; conv1 -> conv2 -> conv3 via WMMA with
// per-wave LDS round-trips for the D->A layout transpose; max over neighbors.
// Weight B-fragments are batch-loaded per output tile so the loads issue as a
// clause and overlap the previous tile's WMMAs.
// ---------------------------------------------------------------------------
__global__ __launch_bounds__(128) void edgeconv_kernel(
    const _Float16* __restrict__ ph, const int* __restrict__ kidx,
    const _Float16* __restrict__ w1p, const _Float16* __restrict__ w2p,
    const _Float16* __restrict__ w3p,
    const float* __restrict__ s1, const float* __restrict__ t1,
    const float* __restrict__ s2, const float* __restrict__ t2,
    const float* __restrict__ b3, _Float16* __restrict__ hmax) {
  __shared__ __align__(16) _Float16 hbuf[4][16 * 256];

  int w    = threadIdx.x >> 5;
  int lane = threadIdx.x & 31;
  int q  = lane & 15;
  int hi = lane >> 4;
  int pt = blockIdx.x * 4 + w;
  int b  = pt >> 12;           // / ND
  int n  = pt & (ND - 1);

  const _Float16* cen = ph + ((size_t)b * ND + n) * CD;
  int nj = kidx[(size_t)pt * KNN + q];   // this lane's neighbor row (M = q)
  const _Float16* nb = ph + ((size_t)b * ND + nj) * CD;

  // Edge features: [16 x 256] = cat(central - neigh, central)
  V16 af[8];
  #pragma unroll
  for (int ks = 0; ks < 4; ++ks) {
    int c0 = ks * 32 + 8 * hi;
    v8h c_lo = *(const v8h*)(cen + c0);
    v8h c_hi = *(const v8h*)(cen + c0 + 16);
    af[ks].h[0]     = c_lo - *(const v8h*)(nb + c0);
    af[ks].h[1]     = c_hi - *(const v8h*)(nb + c0 + 16);
    af[ks + 4].h[0] = c_lo;        // channels 128..255 = central
    af[ks + 4].h[1] = c_hi;
  }

  _Float16* hb = hbuf[w];

  // conv1: [16x256] @ [256x128], fold BN + bias + lrelu
  #pragma unroll
  for (int nt = 0; nt < 8; ++nt) {
    v16h bf[8];
    #pragma unroll
    for (int ks = 0; ks < 8; ++ks)
      bf[ks] = *(const v16h*)(w1p + (((ks * 8 + nt) * 32 + lane) << 4));
    v8f acc = {};
    #pragma unroll
    for (int ks = 0; ks < 8; ++ks) acc = wmma_f16(af[ks].v, bf[ks], acc);
    int co = nt * 16 + q;
    float sc = s1[co], tc = t1[co];
    #pragma unroll
    for (int i = 0; i < 8; ++i) {
      float z = acc[i] * sc + tc;
      z = (z >= 0.f) ? z : 0.2f * z;
      hb[(i + 8 * hi) * 128 + co] = (_Float16)z;
    }
  }
  __syncthreads();

  V16 a2[4];
  #pragma unroll
  for (int ks = 0; ks < 4; ++ks) {
    int c0 = ks * 32 + 8 * hi;
    a2[ks].h[0] = *(const v8h*)(hb + q * 128 + c0);
    a2[ks].h[1] = *(const v8h*)(hb + q * 128 + c0 + 16);
  }
  __syncthreads();

  // conv2: [16x128] @ [128x128]
  #pragma unroll
  for (int nt = 0; nt < 8; ++nt) {
    v16h bf[4];
    #pragma unroll
    for (int ks = 0; ks < 4; ++ks)
      bf[ks] = *(const v16h*)(w2p + (((ks * 8 + nt) * 32 + lane) << 4));
    v8f acc = {};
    #pragma unroll
    for (int ks = 0; ks < 4; ++ks) acc = wmma_f16(a2[ks].v, bf[ks], acc);
    int co = nt * 16 + q;
    float sc = s2[co], tc = t2[co];
    #pragma unroll
    for (int i = 0; i < 8; ++i) {
      float z = acc[i] * sc + tc;
      z = (z >= 0.f) ? z : 0.2f * z;
      hb[(i + 8 * hi) * 128 + co] = (_Float16)z;
    }
  }
  __syncthreads();

  V16 a3[4];
  #pragma unroll
  for (int ks = 0; ks < 4; ++ks) {
    int c0 = ks * 32 + 8 * hi;
    a3[ks].h[0] = *(const v8h*)(hb + q * 128 + c0);
    a3[ks].h[1] = *(const v8h*)(hb + q * 128 + c0 + 16);
  }

  // conv3: [16x128] @ [128x256] + b3, then max over the 16 neighbors
  #pragma unroll
  for (int nt = 0; nt < 16; ++nt) {
    v16h bf[4];
    #pragma unroll
    for (int ks = 0; ks < 4; ++ks)
      bf[ks] = *(const v16h*)(w3p + (((ks * 16 + nt) * 32 + lane) << 4));
    v8f acc = {};
    #pragma unroll
    for (int ks = 0; ks < 4; ++ks) acc = wmma_f16(a3[ks].v, bf[ks], acc);
    int co = nt * 16 + q;
    float bb = b3[co];
    float mx = -3.0e38f;
    #pragma unroll
    for (int i = 0; i < 8; ++i) mx = fmaxf(mx, acc[i] + bb);
    mx = fmaxf(mx, __shfl_xor(mx, 16, 32));   // combine neighbor halves
    if (lane < 16) hmax[(size_t)pt * 256 + co] = (_Float16)mx;
  }
}

// ---------------------------------------------------------------------------
// Final MLP [BN x 256] @ [256 x 256] + bm, fused point-shuffle + transpose:
// out[b, c&127, 2n + (c>>7)] . One wave per (16-point, 16-channel) tile.
// ---------------------------------------------------------------------------
__global__ __launch_bounds__(32) void mlp_kernel(const _Float16* __restrict__ hmax,
                                                 const _Float16* __restrict__ wmp,
                                                 const float* __restrict__ bm,
                                                 float* __restrict__ out) {
  int lane = threadIdx.x;
  int q  = lane & 15;
  int hi = lane >> 4;
  int p0 = blockIdx.x * 16;     // global point tile (never crosses batch)
  int ct = blockIdx.y;          // output channel tile 0..15

  const _Float16* arow = hmax + (size_t)(p0 + q) * 256;
  V16 af[8];
  v16h bf[8];
  #pragma unroll
  for (int ks = 0; ks < 8; ++ks) {
    int c0 = ks * 32 + 8 * hi;
    af[ks].h[0] = *(const v8h*)(arow + c0);
    af[ks].h[1] = *(const v8h*)(arow + c0 + 16);
    bf[ks] = *(const v16h*)(wmp + (((ks * 16 + ct) * 32 + lane) << 4));
  }

  v8f acc = {};
  #pragma unroll
  for (int ks = 0; ks < 8; ++ks) acc = wmma_f16(af[ks].v, bf[ks], acc);

  int c = ct * 16 + q;
  float bias = bm[c];
  int co2 = c & 127;
  int s   = c >> 7;
  #pragma unroll
  for (int i = 0; i < 8; ++i) {
    int pt = p0 + i + 8 * hi;
    int b  = pt >> 12;
    int n  = pt & (ND - 1);
    out[((size_t)(b * 128 + co2)) * (2 * ND) + 2 * n + s] = acc[i] + bias;
  }
}

// ---------------------------------------------------------------------------
extern "C" void kernel_launch(void* const* d_in, const int* in_sizes, int n_in,
                              void* d_out, int out_size, void* d_ws, size_t ws_size,
                              hipStream_t stream) {
  const float* x   = (const float*)d_in[0];
  const float* w1  = (const float*)d_in[1];
  const float* b1  = (const float*)d_in[2];
  const float* g1  = (const float*)d_in[3];
  const float* be1 = (const float*)d_in[4];
  const float* m1  = (const float*)d_in[5];
  const float* v1  = (const float*)d_in[6];
  const float* w2  = (const float*)d_in[7];
  const float* b2  = (const float*)d_in[8];
  const float* g2  = (const float*)d_in[9];
  const float* be2 = (const float*)d_in[10];
  const float* m2  = (const float*)d_in[11];
  const float* v2  = (const float*)d_in[12];
  const float* w3  = (const float*)d_in[13];
  const float* b3  = (const float*)d_in[14];
  const float* wm  = (const float*)d_in[15];
  const float* bm  = (const float*)d_in[16];

  char* ws = (char*)d_ws;
  float*    pf   = (float*)(ws);                 //  8 MiB  p[B,N,C] f32
  _Float16* ph   = (_Float16*)(ws + 8388608);    //  4 MiB  p f16
  float*    sq   = (float*)(ws + 12582912);      // 64 KiB  row norms
  int*      kidx = (int*)(ws + 12648448);        //  1 MiB  kNN indices
  _Float16* hmax = (_Float16*)(ws + 13697024);   //  8 MiB  pooled features f16
  _Float16* w1p  = (_Float16*)(ws + 22085632);   // 64 KiB
  _Float16* w2p  = (_Float16*)(ws + 22151168);   // 32 KiB
  _Float16* w3p  = (_Float16*)(ws + 22183936);   // 64 KiB
  _Float16* wmp  = (_Float16*)(ws + 22249472);   // 128 KiB
  float*    s1   = (float*)(ws + 22380544);
  float*    t1   = s1 + 128;
  float*    s2   = t1 + 128;
  float*    t2   = s2 + 128;

  pack_weights_kernel<<<576, 256, 0, stream>>>(w1, w2, w3, wm, w1p, w2p, w3p, wmp);
  pack_bn_kernel<<<1, 128, 0, stream>>>(b1, g1, be1, m1, v1,
                                        b2, g2, be2, m2, v2, s1, t1, s2, t2);
  transpose_kernel<<<dim3(ND / 256, CD, BD), 256, 0, stream>>>(x, pf, ph);
  sqnorm_kernel<<<(BD * ND) / 8, 256, 0, stream>>>(pf, sq);
  knn_kernel<<<dim3(ND / 16, BD), 32, 0, stream>>>(pf, ph, sq, kidx);
  edgeconv_kernel<<<(BD * ND) / 4, 128, 0, stream>>>(ph, kidx, w1p, w2p, w3p,
                                                     s1, t1, s2, t2, b3, hmax);
  mlp_kernel<<<dim3((BD * ND) / 16, 16), 32, 0, stream>>>(hmax, wmp, bm,
                                                          (float*)d_out);
}
